// DecoderRNN_openattn_85512798863481
// MI455X (gfx1250) — matrix operand order repaired and checked by reading, not compiled, not run on hardware
//
#include <hip/hip_runtime.h>
#include <cstddef>

#define L_   128
#define B_   256
#define IND_ 512
#define H_   1024
#define ENC_ 8
#define OUT_ 512

typedef _Float16 half8  __attribute__((ext_vector_type(8)));
typedef _Float16 half16 __attribute__((ext_vector_type(16)));
typedef float    float8 __attribute__((ext_vector_type(8)));
typedef int      int4v  __attribute__((ext_vector_type(4)));

#if defined(__AMDGCN__) && __has_builtin(__builtin_amdgcn_global_load_async_to_lds_b128)
#define HAVE_ASYNC_LDS 1
#else
#define HAVE_ASYNC_LDS 0
#endif

// 16-byte global->LDS copy: async (ASYNCcnt-tracked, no VGPR staging) if available.
__device__ inline void copy16_g2l(const _Float16* g, _Float16* l) {
#if HAVE_ASYNC_LDS
  __builtin_amdgcn_global_load_async_to_lds_b128((int4v*)g, (int4v*)l, 0, 0);
#else
  *(half8*)l = *(const half8*)g;
#endif
}

__device__ inline void async_drain() {
#if HAVE_ASYNC_LDS
#if __has_builtin(__builtin_amdgcn_s_wait_asynccnt)
  __builtin_amdgcn_s_wait_asynccnt(0);
#else
  asm volatile("s_wait_asynccnt 0x0" ::: "memory");
#endif
#endif
}

// ---------------------------------------------------------------------------
// WMMA GEMM:  C[M,N] = A[M,K](f16) @ W[N,K](f16)^T + bias[N]
// BM=128 x BN=64 x BK=32, 256 threads = 8 waves (wave32), wave grid 4x2,
// each wave 32x32 via 4x v_wmma_f32_16x16x32_f16.
// Double-buffered LDS, async global->LDS staging, 1 barrier per K-step.
// N is a template arg so the epilogue uses immediate-offset stores.
// ---------------------------------------------------------------------------
constexpr int BM = 128, BN = 64, BK = 32;
constexpr int LDSA = 40, LDSB = 40;   // padded strides (halves): conflict-free b128

__device__ inline half16 frag_a(const _Float16* tile, int lane) {
  // ISA 16-bit A 16x32: lanes0-15: e0..7=K0..7, e8..15=K16..23
  //                     lanes16-31: e0..7=K8..15, e8..15=K24..31
  int lr = lane & 15;
  int kh = (lane >> 4) << 3;
  const _Float16* p = tile + lr * LDSA;
  half8 lo = *(const half8*)(p + kh);
  half8 hi = *(const half8*)(p + 16 + kh);
  return __builtin_shufflevector(lo, hi, 0,1,2,3,4,5,6,7,8,9,10,11,12,13,14,15);
}

__device__ inline half16 frag_b(const _Float16* tile, int lane) {
  // ISA 16-bit B 32x16: lanes0-15: e0..7=K0..7, e8..15=K16..23; lanes16-31: e0..7=K8..15, e8..15=K24..31.
  int cn = lane & 15;
  int ks = (lane >> 4) << 3;
  const _Float16* p = tile + cn * LDSB + ks;
  half8 lo = *(const half8*)(p);
  half8 hi = *(const half8*)(p + 16);
  return __builtin_shufflevector(lo, hi, 0,1,2,3,4,5,6,7,8,9,10,11,12,13,14,15);
}

__device__ inline void stage_tiles(const _Float16* __restrict__ A, int lda,
                                   const _Float16* __restrict__ W, int ldb,
                                   int rowBase, int colBase, int k0, int tid,
                                   _Float16* as, _Float16* bs) {
#pragma unroll
  for (int it = 0; it < (BM * BK / 8) / 256; ++it) {   // 2 iters
    int c   = tid + it * 256;
    int row = c >> 2;
    int cc  = (c & 3) << 3;
    copy16_g2l(A + (long)(rowBase + row) * lda + k0 + cc, &as[row * LDSA + cc]);
  }
  {
    int c   = tid;                                     // BN*BK/8 == 256 chunks
    int row = c >> 2;
    int cc  = (c & 3) << 3;
    copy16_g2l(W + (long)(colBase + row) * ldb + k0 + cc, &bs[row * LDSB + cc]);
  }
}

template <int N, bool WF32>
__global__ __launch_bounds__(256)
void wmma_gemm(const _Float16* __restrict__ A, int lda,
               const _Float16* __restrict__ W, int ldb,
               const float* __restrict__ bias,
               float* __restrict__ Cf, _Float16* __restrict__ Ch, int K) {
  __shared__ __align__(16) _Float16 As[2][BM * LDSA];
  __shared__ __align__(16) _Float16 Bs[2][BN * LDSB];

  const int tid  = threadIdx.x;
  const int lane = tid & 31;
  const int wave = tid >> 5;
  const int wm = wave >> 1;             // 0..3
  const int wn = wave & 1;              // 0..1
  const int rowBase = blockIdx.x * BM;
  const int colBase = blockIdx.y * BN;

  float8 acc[2][2];
#pragma unroll
  for (int i = 0; i < 2; ++i)
#pragma unroll
    for (int j = 0; j < 2; ++j) acc[i][j] = (float8)(0.0f);

  int buf = 0;
  stage_tiles(A, lda, W, ldb, rowBase, colBase, 0, tid, As[0], Bs[0]);

  for (int k0 = 0; k0 < K; k0 += BK) {
    async_drain();
    __syncthreads();                    // current buffer complete in LDS
    if (k0 + BK < K)
      stage_tiles(A, lda, W, ldb, rowBase, colBase, k0 + BK, tid,
                  As[buf ^ 1], Bs[buf ^ 1]);

    half16 a0 = frag_a(&As[buf][(wm * 32 + 0)  * LDSA], lane);
    half16 a1 = frag_a(&As[buf][(wm * 32 + 16) * LDSA], lane);
    half16 b0 = frag_b(&Bs[buf][(wn * 32 + 0)  * LDSB], lane);
    half16 b1 = frag_b(&Bs[buf][(wn * 32 + 16) * LDSB], lane);

    acc[0][0] = __builtin_amdgcn_wmma_f32_16x16x32_f16(false, a0, false, b0, (short)0, acc[0][0], false, false);
    acc[0][1] = __builtin_amdgcn_wmma_f32_16x16x32_f16(false, a0, false, b1, (short)0, acc[0][1], false, false);
    acc[1][0] = __builtin_amdgcn_wmma_f32_16x16x32_f16(false, a1, false, b0, (short)0, acc[1][0], false, false);
    acc[1][1] = __builtin_amdgcn_wmma_f32_16x16x32_f16(false, a1, false, b1, (short)0, acc[1][1], false, false);
    buf ^= 1;
  }

  // epilogue: elem v -> row = v + (lane>=16 ? 8 : 0), col = lane%16
#pragma unroll
  for (int i = 0; i < 2; ++i) {
    int rbase = rowBase + wm * 32 + i * 16 + ((lane >> 4) << 3);
#pragma unroll
    for (int j = 0; j < 2; ++j) {
      int col   = colBase + wn * 32 + j * 16 + (lane & 15);
      float bv  = bias[col];
      long base = (long)rbase * N + col;
      if constexpr (WF32) {
        float* cp = Cf + base;
#pragma unroll
        for (int v = 0; v < 8; ++v) cp[v * N] = acc[i][j][v] + bv;
      } else {
        _Float16* cp = Ch + base;
#pragma unroll
        for (int v = 0; v < 8; ++v) cp[v * N] = (_Float16)(acc[i][j][v] + bv);
      }
    }
  }
}

// ---------------------------------------------------------------------------
// GRU gate fusion:  r,z = sigmoid(gi+gh); n = tanh(gi_n + r*gh_n);
//                   h' = (1-z)*n + z*h   -> hout (f32) and ys[t] (f16)
// ---------------------------------------------------------------------------
__global__ __launch_bounds__(256)
void gru_gate(const _Float16* __restrict__ GI, const float* __restrict__ GH,
              const float* __restrict__ hin, float* __restrict__ hout,
              _Float16* __restrict__ ys) {
  int i = blockIdx.x * blockDim.x + threadIdx.x;   // over B*H
  if (i >= B_ * H_) return;
  int b = i / H_, j = i % H_;
  long g = (long)b * 3 * H_ + j;
  float r  = (float)GI[g]          + GH[g];
  float z  = (float)GI[g + H_]     + GH[g + H_];
  float nn = (float)GI[g + 2 * H_];
  float hn = GH[g + 2 * H_];
  r = 1.0f / (1.0f + __expf(-r));
  z = 1.0f / (1.0f + __expf(-z));
  float n = tanhf(nn + r * hn);
  float h = (1.0f - z) * n + z * hin[i];
  hout[i] = h;
  ys[i]   = (_Float16)h;
}

// ---------------------------------------------------------------------------
template <typename T>
__global__ __launch_bounds__(256)
void rowdot(const T* __restrict__ X, const float* __restrict__ w,
            float* __restrict__ s, int R) {
  int r    = blockIdx.x * 8 + (threadIdx.x >> 5);
  int lane = threadIdx.x & 31;
  if (r >= R) return;
  const T* row = X + (long)r * H_;
  float acc = 0.0f;
  for (int h = lane; h < H_; h += 32) acc += (float)row[h] * w[h];
#pragma unroll
  for (int off = 16; off; off >>= 1) acc += __shfl_xor(acc, off, 32);
  if (lane == 0) s[r] = acc;
}

// ---------------------------------------------------------------------------
__global__ __launch_bounds__(256)
void attn_ctx(const float* __restrict__ s_out, const float* __restrict__ s_enc,
              const float* __restrict__ ba, const float* __restrict__ enc,
              const _Float16* __restrict__ ys1, _Float16* __restrict__ cat) {
  int r = blockIdx.x;                   // r = b*L + l
  int b = r / L_, l = r % L_;
  float so = s_out[l * B_ + b];
  float w[ENC_];
  float mx = -1e30f;
#pragma unroll
  for (int j = 0; j < ENC_; ++j) {
    float e = tanhf(so + s_enc[j * B_ + b] + ba[0]);
    w[j] = e;
    mx = fmaxf(mx, e);
  }
  float sum = 0.0f;
#pragma unroll
  for (int j = 0; j < ENC_; ++j) { w[j] = __expf(w[j] - mx); sum += w[j]; }
  float inv = 1.0f / sum;
  for (int h = threadIdx.x; h < H_; h += blockDim.x) {
    float c = 0.0f;
#pragma unroll
    for (int j = 0; j < ENC_; ++j)
      c += w[j] * inv * enc[((long)j * B_ + b) * H_ + h];
    cat[(long)r * 2 * H_ + h]      = (_Float16)c;
    cat[(long)r * 2 * H_ + H_ + h] = ys1[((long)l * B_ + b) * H_ + h];
  }
}

// ---------------------------------------------------------------------------
__global__ void scale_f32_to_f16(const float* __restrict__ s, _Float16* __restrict__ d,
                                 long n, float alpha) {
  for (long i = blockIdx.x * (long)blockDim.x + threadIdx.x; i < n;
       i += (long)gridDim.x * blockDim.x)
    d[i] = (_Float16)(s[i] * alpha);
}

__global__ void copy_hidden(const float* __restrict__ h0, const float* __restrict__ h1,
                            float* __restrict__ dst, int n) {
  int i = blockIdx.x * blockDim.x + threadIdx.x;
  if (i < n) { dst[i] = h0[i]; dst[n + i] = h1[i]; }
}

// ---------------------------------------------------------------------------
extern "C" void kernel_launch(void* const* d_in, const int* in_sizes, int n_in,
                              void* d_out, int out_size, void* d_ws, size_t ws_size,
                              hipStream_t stream) {
  (void)in_sizes; (void)n_in; (void)out_size; (void)ws_size;
  const float* input_ = (const float*)d_in[0];
  const float* hidden = (const float*)d_in[1];
  // d_in[2] input_lengths: unused by the reference math
  const float* enc    = (const float*)d_in[3];
  const float* W_ih0  = (const float*)d_in[4];
  const float* W_hh0  = (const float*)d_in[5];
  const float* b_ih0  = (const float*)d_in[6];
  const float* b_hh0  = (const float*)d_in[7];
  const float* W_ih1  = (const float*)d_in[8];
  const float* W_hh1  = (const float*)d_in[9];
  const float* b_ih1  = (const float*)d_in[10];
  const float* b_hh1  = (const float*)d_in[11];
  const float* Wa     = (const float*)d_in[12];
  const float* ba     = (const float*)d_in[13];
  const float* W_out  = (const float*)d_in[14];
  const float* b_out  = (const float*)d_in[15];
  float* out = (float*)d_out;

  // ---- workspace carve-out (256B aligned) ----
  char* p = (char*)d_ws;
  auto carve = [&](size_t bytes) -> char* {
    char* r = p;
    p += (bytes + 255) & ~(size_t)255;
    return r;
  };
  const size_t RB = (size_t)L_ * B_;            // 32768 rows
  _Float16* wih0h = (_Float16*)carve((size_t)3 * H_ * IND_ * 2);
  _Float16* whh0h = (_Float16*)carve((size_t)3 * H_ * H_ * 2);
  _Float16* wih1h = (_Float16*)carve((size_t)3 * H_ * H_ * 2);
  _Float16* whh1h = (_Float16*)carve((size_t)3 * H_ * H_ * 2);
  _Float16* wouth = (_Float16*)carve((size_t)OUT_ * 2 * H_ * 2);
  _Float16* x16   = (_Float16*)carve(RB * IND_ * 2);
  _Float16* hid16 = (_Float16*)carve((size_t)2 * B_ * H_ * 2);
  _Float16* gi0   = (_Float16*)carve(RB * 3 * H_ * 2);
  _Float16* gi1   = (_Float16*)carve(RB * 3 * H_ * 2);
  _Float16* ys0   = (_Float16*)carve(RB * H_ * 2);
  _Float16* ys1   = (_Float16*)carve(RB * H_ * 2);
  _Float16* cat   = (_Float16*)carve(RB * 2 * H_ * 2);
  float*    gh    = (float*)carve((size_t)B_ * 3 * H_ * 4);
  float*    h0a   = (float*)carve((size_t)B_ * H_ * 4);
  float*    h0b   = (float*)carve((size_t)B_ * H_ * 4);
  float*    h1a   = (float*)carve((size_t)B_ * H_ * 4);
  float*    h1b   = (float*)carve((size_t)B_ * H_ * 4);
  float*    sout  = (float*)carve(RB * 4);
  float*    senc  = (float*)carve((size_t)ENC_ * B_ * 4);

  dim3 blk(256);

  // 1) f32 -> f16 staging (weights, scaled input, initial hidden)
  scale_f32_to_f16<<<512, blk, 0, stream>>>(W_ih0, wih0h, (long)3 * H_ * IND_, 1.0f);
  scale_f32_to_f16<<<512, blk, 0, stream>>>(W_hh0, whh0h, (long)3 * H_ * H_, 1.0f);
  scale_f32_to_f16<<<512, blk, 0, stream>>>(W_ih1, wih1h, (long)3 * H_ * H_, 1.0f);
  scale_f32_to_f16<<<512, blk, 0, stream>>>(W_hh1, whh1h, (long)3 * H_ * H_, 1.0f);
  scale_f32_to_f16<<<512, blk, 0, stream>>>(W_out, wouth, (long)OUT_ * 2 * H_, 1.0f);
  scale_f32_to_f16<<<512, blk, 0, stream>>>(input_, x16, (long)RB * IND_, 0.9f);
  scale_f32_to_f16<<<512, blk, 0, stream>>>(hidden, hid16, (long)2 * B_ * H_, 1.0f);

  // 2) GI0 = x16 @ W_ih0^T + b_ih0   (batched over all timesteps)
  wmma_gemm<3 * H_, false><<<dim3(RB / BM, (3 * H_) / BN), blk, 0, stream>>>(
      x16, IND_, wih0h, IND_, b_ih0, nullptr, gi0, IND_);

  // 3) GRU layer 0 (sequential; A = f16 h from previous step)
  const float* hin = hidden;                 // f32 hidden[0] for gate math
  float* hping = h0a; float* hpong = h0b;
  for (int t = 0; t < L_; ++t) {
    const _Float16* a16 = (t == 0) ? hid16 : ys0 + (size_t)(t - 1) * B_ * H_;
    wmma_gemm<3 * H_, true><<<dim3(B_ / BM, (3 * H_) / BN), blk, 0, stream>>>(
        a16, H_, whh0h, H_, b_hh0, gh, nullptr, H_);
    gru_gate<<<dim3((B_ * H_) / 256), blk, 0, stream>>>(
        gi0 + (size_t)t * B_ * 3 * H_, gh, hin, hping, ys0 + (size_t)t * B_ * H_);
    hin = hping;
    { float* tmp = hping; hping = hpong; hpong = tmp; }
  }
  const float* hT0 = hin;

  // 4) GI1 = ys0 @ W_ih1^T + b_ih1  (batched)
  wmma_gemm<3 * H_, false><<<dim3(RB / BM, (3 * H_) / BN), blk, 0, stream>>>(
      ys0, H_, wih1h, H_, b_ih1, nullptr, gi1, H_);

  // 5) GRU layer 1
  hin = hidden + (size_t)B_ * H_;
  hping = h1a; hpong = h1b;
  for (int t = 0; t < L_; ++t) {
    const _Float16* a16 = (t == 0) ? hid16 + (size_t)B_ * H_
                                   : ys1 + (size_t)(t - 1) * B_ * H_;
    wmma_gemm<3 * H_, true><<<dim3(B_ / BM, (3 * H_) / BN), blk, 0, stream>>>(
        a16, H_, whh1h, H_, b_hh1, gh, nullptr, H_);
    gru_gate<<<dim3((B_ * H_) / 256), blk, 0, stream>>>(
        gi1 + (size_t)t * B_ * 3 * H_, gh, hin, hping, ys1 + (size_t)t * B_ * H_);
    hin = hping;
    { float* tmp = hping; hping = hpong; hpong = tmp; }
  }
  const float* hT1 = hin;

  // 6) attention scalars
  rowdot<_Float16><<<dim3(RB / 8), blk, 0, stream>>>(ys1, Wa, sout, (int)RB);
  rowdot<float><<<dim3((ENC_ * B_) / 8), blk, 0, stream>>>(enc, Wa + H_, senc, ENC_ * B_);

  // 7) softmax + context + concat rows [b*L+l][2H]
  attn_ctx<<<dim3(RB), blk, 0, stream>>>(sout, senc, ba, enc, ys1, cat);

  // 8) out[B*L, OUT] = cat @ W_out^T + b_out  -> d_out head
  wmma_gemm<OUT_, true><<<dim3(RB / BM, OUT_ / BN), blk, 0, stream>>>(
      cat, 2 * H_, wouth, 2 * H_, b_out, out, nullptr, 2 * H_);

  // 9) hidden_out [2,B,H] -> d_out tail
  copy_hidden<<<dim3((B_ * H_) / 256), blk, 0, stream>>>(
      hT0, hT1, out + (size_t)B_ * L_ * OUT_, B_ * H_);
}
